// classSR_3class_fused_fsrcnn_net_7533372637950
// MI455X (gfx1250) — matrix operand order repaired and checked by reading, not compiled, not run on hardware
//
#include <hip/hip_runtime.h>
#include <stdint.h>

typedef _Float16 half_t;
typedef __attribute__((ext_vector_type(16))) _Float16 v16h;
typedef __attribute__((ext_vector_type(8)))  _Float16 v8h;
typedef __attribute__((ext_vector_type(8)))  float    v8f;

#define DEV __device__ __forceinline__

// ---------------- workspace layout (units: f16 halves) ----------------
#define OFF_W0   0                       // cls conv0 Bt [128][64] (K=48 used)
#define OFF_W1   (OFF_W0 + 128*64)       // cls 1x1 Bt [128][128]
#define OFF_W2   (OFF_W1 + 128*128)
#define OFF_W3   (OFF_W2 + 128*128)
#define OFF_W4   (OFF_W3 + 128*128)      // [32][128]
#define OFF_BR   (OFF_W4 + 32*128)       // per-branch expert weights
#define WS_HALVES_CLS_BYTES (OFF_BR * 2) // 122880 B of classifier weights
// per-branch internal layout (halves):
//   head: 10 pair-units [16][32] (u = tx*2+pi, ty = pi*2 + k/16)  = 5120
//         5 single-units [16][16] (ty=4)                          = 1280
#define BR_HEAD  0
#define BR_B0    6400                    // [16][16]
//   map (per layer, 2304): 3 pair-units [16][32] (tx, ty=k/16), then
//                          3 single-units [16][16] (ty=2)
#define BR_MAP   6656
#define BR_B6    15872                   // [16][16]
#define BR_SZ    16128
#define WS_HALVES (OFF_BR + 3*BR_SZ)     // 109824 halves
#define OFF_ROUTE_BYTES (WS_HALVES*2)

// ---------------- async global->LDS staging (CDNA5 ASYNCcnt path) ------------
DEV void async_copy(void* ldsDst, const void* gSrc, int bytes, int tid) {
  uint32_t l = (uint32_t)(uintptr_t)ldsDst;      // LDS aperture: low 32 bits
  uint64_t g = (uint64_t)(uintptr_t)gSrc;
  for (int i = tid * 16; i < bytes; i += 256 * 16) {
    asm volatile("global_load_async_to_lds_b128 %0, %1, off"
                 :: "v"(l + (uint32_t)i), "v"(g + (uint64_t)i)
                 : "memory");
  }
}
DEV void async_wait() { asm volatile("s_wait_asynccnt 0x0" ::: "memory"); }

// ---------------- WMMA fragment loaders (wave32 layouts per CDNA5 ISA) -------
// A (16-bit 16x32): lane<16 -> M=lane, K0..7 / K16..23 ; lane>=16 -> M=lane-16,
// K8..15 / K24..31.  Effective-K16: rows 16 halves wide, upper K zero.
DEV v16h frag_a16(const half_t* row0, int rowStride) {
  const int lane = threadIdx.x & 31;
  const int m = lane & 15, kb = (lane >> 4) << 3;
  v8h lo = *(const v8h*)(row0 + m * rowStride + kb);
  v16h a;
#pragma unroll
  for (int i = 0; i < 8; ++i) a[i] = lo[i];
#pragma unroll
  for (int i = 8; i < 16; ++i) a[i] = (half_t)0;
  return a;
}

// Pair-tap A: K0..15 = 16ch at pixel row r, K16..31 = 16ch at pixel row r+1.
DEV v16h frag_a_pair(const half_t* row0, int pitchHalves) {
  const int lane = threadIdx.x & 31;
  const int m = lane & 15, kb = (lane >> 4) << 3;
  const half_t* p = row0 + m * 16 + kb;
  v8h lo = *(const v8h*)p;
  v8h hi = *(const v8h*)(p + pitchHalves);
  v16h a;
#pragma unroll
  for (int i = 0; i < 8; ++i) { a[i] = lo[i]; a[i + 8] = hi[i]; }
  return a;
}

// B (16-bit 32x16): lane<16 -> N=lane K0..7 / K16..23 ; lane>=16 -> N=lane-16 K8..15 / K24..31.
DEV v16h frag_b16(const half_t* w) {   // [16][16], K16..31 = 0
  const int lane = threadIdx.x & 31;
  v16h b;
  v8h x0 = *(const v8h*)(w + (lane & 15) * 16 + ((lane >> 4) << 3));
#pragma unroll
  for (int i = 0; i < 8; ++i) { b[i] = x0[i]; b[i + 8] = (half_t)0; }
  return b;
}

DEV v16h frag_a32(const half_t* X, int row0, int rowStride, int kcol) {
  const int lane = threadIdx.x & 31;
  const int m = lane & 15, kb = (lane >> 4) << 3;
  const half_t* base = X + (row0 + m) * rowStride + kcol;
  v8h lo = *(const v8h*)(base + kb);
  v8h hi = *(const v8h*)(base + 16 + kb);
  v16h a;
#pragma unroll
  for (int i = 0; i < 8; ++i) { a[i] = lo[i]; a[i + 8] = hi[i]; }
  return a;
}

DEV v16h frag_b32(const half_t* W, int n0, int rowStride, int kcol) {
  const int lane = threadIdx.x & 31;
  const int n = lane & 15, ks = (lane >> 4) << 3;
  const half_t* base = W + (n0 + n) * rowStride + kcol + ks;
  v8h x0 = *(const v8h*)base;
  v8h x1 = *(const v8h*)(base + 16);
  v16h b;
#pragma unroll
  for (int i = 0; i < 8; ++i) { b[i] = x0[i]; b[i + 8] = x1[i]; }
  return b;
}

DEV v8f wmma_f32(v16h a, v16h b, v8f c) {
  return __builtin_amdgcn_wmma_f32_16x16x32_f16(false, a, false, b, (short)0, c,
                                                false, false);
}

// D layout: lane<16 -> N=lane, M=r ; lane>=16 -> N=lane-16, M=8+r.
DEV void store_tile(half_t* dst, int y, int x0, const v8f& acc,
                    const float* biasv, const float* preluv, bool doPrelu,
                    int lane) {
  const int n = lane & 15, mb = (lane >> 4) << 3;
  const float bv = biasv[n], av = preluv[n];
#pragma unroll
  for (int r = 0; r < 8; ++r) {
    float v = acc[r] + bv;
    if (doPrelu) v = v >= 0.f ? v : av * v;
    dst[((y + 1) * 34 + x0 + 1 + mb + r) * 16 + n] = (half_t)v;
  }
}

// ---------------- kernel 1: weight prep (f32 -> padded/packed f16) -----------
__global__ void prep_weights(const float* __restrict__ cw0, const float* __restrict__ cw1,
                             const float* __restrict__ cw2, const float* __restrict__ cw3,
                             const float* __restrict__ cw4, const float* __restrict__ hw,
                             const float* __restrict__ b0w, const float* __restrict__ mw,
                             const float* __restrict__ b6w, half_t* __restrict__ ws) {
  for (int idx = blockIdx.x * blockDim.x + threadIdx.x; idx < WS_HALVES;
       idx += gridDim.x * blockDim.x) {
    float v = 0.f;
    if (idx < OFF_W1) {
      int n = idx >> 6, k = idx & 63;
      if (k < 48) v = cw0[n * 48 + k];                   // [128][3][4][4]
    } else if (idx < OFF_W2) { v = cw1[idx - OFF_W1];
    } else if (idx < OFF_W3) { v = cw2[idx - OFF_W2];
    } else if (idx < OFF_W4) { v = cw3[idx - OFF_W3];
    } else if (idx < OFF_BR) { v = cw4[idx - OFF_W4];
    } else {
      int rel = idx - OFF_BR;
      int br = rel / BR_SZ, r2 = rel % BR_SZ;
      if (r2 < BR_B0) {                                  // head 5x5 (pair-packed)
        int n, c, ty, tx;
        if (r2 < 5120) {
          int u = r2 >> 9; tx = u >> 1; int pi = u & 1;
          int rem = r2 & 511; n = rem >> 5; int k = rem & 31;
          ty = pi * 2 + (k >> 4); c = k & 15;
        } else {
          int rl = r2 - 5120; tx = rl >> 8; int nk = rl & 255;
          n = nk >> 4; c = nk & 15; ty = 4;
        }
        if (c < 3) v = hw[((((br * 16 + n) * 3 + c) * 5 + ty) * 5) + tx];
      } else if (r2 < BR_MAP) {                          // b0 1x1
        int nk = r2 - BR_B0, n = nk >> 4, c = nk & 15;
        if (n < 12) v = b0w[(br * 12 + n) * 16 + c];
      } else if (r2 < BR_B6) {                           // map 3x3 (pair-packed)
        int r3 = r2 - BR_MAP;
        int layer = r3 / 2304, r4 = r3 % 2304;
        int n, c, ty, tx;
        if (r4 < 1536) {
          tx = r4 / 512; int rem = r4 & 511; n = rem >> 5; int k = rem & 31;
          ty = k >> 4; c = k & 15;
        } else {
          int rl = r4 - 1536; tx = rl >> 8; int nk = rl & 255;
          n = nk >> 4; c = nk & 15; ty = 2;
        }
        if (n < 12 && c < 12)
          v = mw[(((((br * 4 + layer) * 12 + n) * 12 + c) * 3 + ty) * 3) + tx];
      } else {                                           // b6 1x1
        int nk = r2 - BR_B6, n = nk >> 4, c = nk & 15;
        if (c < 12) v = b6w[(br * 16 + n) * 12 + c];
      }
    }
    ws[idx] = (half_t)v;
  }
}

// ---------------- kernel 2: classifier + top-1 router (1 WG / image) ---------
#define CLS_A0  0
#define CLS_W   8192                     // all cls weights, 122880 B resident
#define CLS_X0  (CLS_W + 122880)
#define CLS_X1  (CLS_X0 + 16384)
#define CLS_C4  (CLS_X1 + 16384)
#define CLS_PL  (CLS_C4 + 8192)
#define SMEM_CLS (CLS_PL + 128)

DEV void gemm128(const half_t* Xin, const half_t* Wl, const float* __restrict__ bias,
                 half_t* Xout, int wave, int lane) {
  for (int t = wave; t < 32; t += 8) {
    int mt = t & 3, nt = t >> 2;
    v8f acc = {};
#pragma unroll
    for (int kk = 0; kk < 4; ++kk)
      acc = wmma_f32(frag_a32(Xin, mt * 16, 128, kk * 32),
                     frag_b32(Wl, nt * 16, 128, kk * 32), acc);
    int n = lane & 15, mb = (lane >> 4) << 3;
    float bv = bias[nt * 16 + n];
#pragma unroll
    for (int r = 0; r < 8; ++r) {
      float v = acc[r] + bv;
      v = v >= 0.f ? v : 0.1f * v;
      Xout[(mt * 16 + mb + r) * 128 + nt * 16 + n] = (half_t)v;
    }
  }
}

__global__ __launch_bounds__(256) void cls_kernel(
    const float* __restrict__ x, const half_t* __restrict__ ws,
    const float* __restrict__ cb0, const float* __restrict__ cb1,
    const float* __restrict__ cb2, const float* __restrict__ cb3,
    const float* __restrict__ cb4, const float* __restrict__ lw,
    const float* __restrict__ lb, int* __restrict__ route) {
  extern __shared__ char smem[];
  half_t* A0 = (half_t*)(smem + CLS_A0);
  half_t* W  = (half_t*)(smem + CLS_W);
  half_t* X0 = (half_t*)(smem + CLS_X0);
  half_t* X1 = (half_t*)(smem + CLS_X1);
  float*  C4 = (float*)(smem + CLS_C4);
  float*  pooled = (float*)(smem + CLS_PL);

  const int tid = threadIdx.x, wave = tid >> 5, lane = tid & 31;
  const int b = blockIdx.x;
  const float* xb = x + (size_t)b * 3072;

  // stage ALL classifier weights LDS-resident via async DMA (no VGPR round trip)
  async_copy(W, ws + OFF_W0, WS_HALVES_CLS_BYTES, tid);

  // im2col for 4x4 stride-4 conv: 64 patches x (48 -> pad 64)
  for (int i = tid; i < 64 * 64; i += 256) {
    int p = i >> 6, k = i & 63;
    float v = 0.f;
    if (k < 48) {
      int c = k >> 4, r = k & 15, iy = r >> 2, ix = r & 3;
      int py = p >> 3, px = p & 7;
      v = xb[(c * 32 + (py * 4 + iy)) * 32 + px * 4 + ix];
    }
    A0[i] = (half_t)v;
  }
  async_wait();
  __syncthreads();

  // layer0: M=64, N=128, K=64 (weights at W + OFF_W0 == W)
  for (int t = wave; t < 32; t += 8) {
    int mt = t & 3, nt = t >> 2;
    v8f acc = {};
#pragma unroll
    for (int kk = 0; kk < 2; ++kk)
      acc = wmma_f32(frag_a32(A0, mt * 16, 64, kk * 32),
                     frag_b32(W, nt * 16, 64, kk * 32), acc);
    int n = lane & 15, mb = (lane >> 4) << 3;
    float bv = cb0[nt * 16 + n];
#pragma unroll
    for (int r = 0; r < 8; ++r) {
      float v = acc[r] + bv;
      v = v >= 0.f ? v : 0.1f * v;
      X0[(mt * 16 + mb + r) * 128 + nt * 16 + n] = (half_t)v;
    }
  }
  __syncthreads();

  const half_t* wsrc[3] = { W + OFF_W1, W + OFF_W2, W + OFF_W3 };
  const float* bsrc[3] = { cb1, cb2, cb3 };
  half_t* bufs[2] = { X0, X1 };
  for (int L = 0; L < 3; ++L) {
    gemm128(bufs[L & 1], wsrc[L], bsrc[L], bufs[(L + 1) & 1], wave, lane);
    __syncthreads();
  }
  { // layer4: N=32, input X1
    int mt = wave & 3, nt = wave >> 2;
    v8f acc = {};
#pragma unroll
    for (int kk = 0; kk < 4; ++kk)
      acc = wmma_f32(frag_a32(X1, mt * 16, 128, kk * 32),
                     frag_b32(W + OFF_W4, nt * 16, 128, kk * 32), acc);
    int n = lane & 15, mb = (lane >> 4) << 3;
    float bv = cb4[nt * 16 + n];
#pragma unroll
    for (int r = 0; r < 8; ++r)
      C4[(mt * 16 + mb + r) * 32 + nt * 16 + n] = acc[r] + bv;
  }
  __syncthreads();
  if (tid < 32) {
    float s = 0.f;
    for (int r = 0; r < 64; ++r) s += C4[r * 32 + tid];
    pooled[tid] = s * (1.f / 64.f);
  }
  __syncthreads();
  if (tid == 0) {
    float best = -3.4e38f; int bi = 0;
    for (int j = 0; j < 3; ++j) {
      float s = lb[j];
      for (int k = 0; k < 32; ++k) s += pooled[k] * lw[j * 32 + k];
      if (s > best) { best = s; bi = j; }
    }
    route[b] = bi;
  }
}

// ---------------- expert conv stages (pair-packed K) -------------------------
DEV void conv5x5head(const half_t* src, half_t* dst, const half_t* w,
                     const float* biasv, const float* preluv, int wave, int lane) {
  for (int t = wave; t < 64; t += 8) {
    const int y = t >> 1, x0 = (t & 1) << 4;
    v8f acc = {};
#pragma unroll
    for (int u = 0; u < 10; ++u) {                       // row-pairs (ty,ty+1)
      const int tx = u >> 1, pi = u & 1;
      v16h a = frag_a_pair(src + ((y + pi * 2) * 36 + x0 + tx) * 16, 576);
      acc = wmma_f32(a, frag_b32(w + u * 512, 0, 32, 0), acc);
    }
#pragma unroll
    for (int tx = 0; tx < 5; ++tx) {                     // ty=4 singles
      v16h a = frag_a16(src + ((y + 4) * 36 + x0 + tx) * 16, 16);
      acc = wmma_f32(a, frag_b16(w + 5120 + tx * 256), acc);
    }
    store_tile(dst, y, x0, acc, biasv, preluv, true, lane);
  }
}

DEV void conv3x3(const half_t* src, half_t* dst, const half_t* w,
                 const float* biasv, const float* preluv, bool doPrelu,
                 int wave, int lane) {
  for (int t = wave; t < 64; t += 8) {
    const int y = t >> 1, x0 = (t & 1) << 4;
    v8f acc = {};
#pragma unroll
    for (int tx = 0; tx < 3; ++tx) {                     // row-pair (ty=0,1)
      v16h a = frag_a_pair(src + (y * 34 + x0 + tx) * 16, 544);
      acc = wmma_f32(a, frag_b32(w + tx * 512, 0, 32, 0), acc);
    }
#pragma unroll
    for (int tx = 0; tx < 3; ++tx) {                     // ty=2 singles
      v16h a = frag_a16(src + ((y + 2) * 34 + x0 + tx) * 16, 16);
      acc = wmma_f32(a, frag_b16(w + 1536 + tx * 256), acc);
    }
    store_tile(dst, y, x0, acc, biasv, preluv, doPrelu, lane);
  }
}

DEV void conv1x1(const half_t* srcInt, half_t* dst, const half_t* w,
                 const float* biasv, const float* preluv, int wave, int lane) {
  for (int t = wave; t < 64; t += 8) {
    const int y = t >> 1, x0 = (t & 1) << 4;
    v8f acc = {};
    v16h a = frag_a16(srcInt + (y * 34 + x0) * 16, 16);
    acc = wmma_f32(a, frag_b16(w), acc);
    store_tile(dst, y, x0, acc, biasv, preluv, false, lane);
  }
}

// ---------------- kernel 3: routed FSRCNN expert (1 WG / image) --------------
#define BR_XH    0                                   // [36*36][16] f16, pad=2
#define BR_ACTA  41472                               // [34*34][16] f16, pad=1
#define BR_ACTB  (BR_ACTA + 36992)
#define BR_W     (BR_ACTB + 36992)                   // all expert weights resident
#define BR_WTAIL (BR_W + 32256)                      // deconv f32 [16][3][9][9]
#define BR_BIAS  (BR_W + 47872)
#define SMEM_BR  (BR_BIAS + 640)

__global__ __launch_bounds__(256) void branch_kernel(
    const float* __restrict__ x, const half_t* __restrict__ ws,
    const int* __restrict__ route,
    const float* __restrict__ hb, const float* __restrict__ ha,
    const float* __restrict__ b0b, const float* __restrict__ mbv,
    const float* __restrict__ ba, const float* __restrict__ b6b,
    const float* __restrict__ tw, const float* __restrict__ tb,
    float* __restrict__ out) {
  extern __shared__ char smem[];
  half_t* xh    = (half_t*)(smem + BR_XH);
  half_t* actA  = (half_t*)(smem + BR_ACTA);
  half_t* actB  = (half_t*)(smem + BR_ACTB);
  half_t* wAll  = (half_t*)(smem + BR_W);
  float*  wTail = (float*)(smem + BR_WTAIL);
  float*  biasH   = (float*)(smem + BR_BIAS);
  float*  preluH  = biasH + 16;
  float*  biasB0  = preluH + 16;
  float*  preluBd = biasB0 + 16;
  float*  biasMp  = preluBd + 16;        // [4][16]
  float*  biasB6  = biasMp + 64;
  float*  tbv     = biasB6 + 16;

  const int tid = threadIdx.x, wave = tid >> 5, lane = tid & 31;
  const int b = blockIdx.x;
  const int r = route[b];
  const half_t* wbr = ws + OFF_BR + r * BR_SZ;

  // stage ALL expert weights (f16 GEMM weights + f32 deconv taps) via async DMA
  async_copy(wAll, wbr, BR_SZ * 2, tid);               // 32256 B
  async_copy(wTail, tw + r * 3888, 3888 * 4, tid);     // 15552 B

  { // zero activation buffers (borders + padded channels must stay 0)
    uint32_t* z = (uint32_t*)smem;
    for (int i = tid; i < (41472 + 2 * 36992) / 4; i += 256) z[i] = 0u;
  }
  __syncthreads();
  const float* xb = x + (size_t)b * 3072;
  for (int i = tid; i < 3072; i += 256) {
    int c = i >> 10, p = i & 1023, y = p >> 5, xc = p & 31;
    xh[((y + 2) * 36 + xc + 2) * 16 + c] = (half_t)xb[i];
  }
  if (tid < 16) {
    biasH[tid]  = hb[r * 16 + tid];
    preluH[tid] = ha[r * 16 + tid];
    biasB0[tid] = (tid < 12) ? b0b[r * 12 + tid] : 0.f;
    preluBd[tid] = (tid < 12) ? ba[r * 12 + tid] : 0.f;
    biasB6[tid] = b6b[r * 16 + tid];
#pragma unroll
    for (int l = 0; l < 4; ++l)
      biasMp[l * 16 + tid] = (tid < 12) ? mbv[(r * 4 + l) * 12 + tid] : 0.f;
  }
  if (tid == 0) { tbv[0] = tb[r * 3]; tbv[1] = tb[r * 3 + 1]; tbv[2] = tb[r * 3 + 2]; }
  async_wait();
  __syncthreads();

  conv5x5head(xh, actA, wAll + BR_HEAD, biasH, preluH, wave, lane);
  __syncthreads();
  conv1x1(actA + 35 * 16, actB, wAll + BR_B0, biasB0, biasB0, wave, lane);
  __syncthreads();
  half_t* src = actB; half_t* dst = actA;
  for (int layer = 0; layer < 4; ++layer) {
    conv3x3(src, dst, wAll + BR_MAP + layer * 2304, biasMp + layer * 16,
            preluBd, layer == 3, wave, lane);
    __syncthreads();
    half_t* t2 = src; src = dst; dst = t2;
  }
  conv1x1(src + 35 * 16, dst, wAll + BR_B6, biasB6, biasB6, wave, lane);
  __syncthreads();
  const half_t* hfin = dst;

  // deconv ConvTranspose2d(16,3,k=9,s=4,p=3,op=1): phase-sparse taps, f32 VALU
  float* ob = out + (size_t)b * 3 * 128 * 128;
  for (int idx = tid; idx < 3 * 128 * 128; idx += 256) {
    int co = idx >> 14, rem = idx & 16383, oy = rem >> 7, ox = rem & 127;
    float s = tbv[co];
    int ky0 = (5 - oy) & 3, kx0 = (5 - ox) & 3;
    for (int ky = ky0; ky < 9; ky += 4) {
      int t = oy + ky - 5; if (t < 0) continue; int iy = t >> 2; if (iy >= 32) continue;
      for (int kx = kx0; kx < 9; kx += 4) {
        int u = ox + kx - 5; if (u < 0) continue; int ix = u >> 2; if (ix >= 32) continue;
        const half_t* hp = hfin + ((iy + 1) * 34 + ix + 1) * 16;
        const float* wp = wTail + co * 81 + (8 - ky) * 9 + (8 - kx);
        float a2 = 0.f;
#pragma unroll
        for (int ci = 0; ci < 16; ++ci) a2 += (float)hp[ci] * wp[ci * 243];
        s += a2;
      }
    }
    ob[idx] = s;
  }
}

// ---------------- launch ----------------
extern "C" void kernel_launch(void* const* d_in, const int* in_sizes, int n_in,
                              void* d_out, int out_size, void* d_ws, size_t ws_size,
                              hipStream_t stream) {
  (void)in_sizes; (void)n_in; (void)out_size; (void)ws_size;
  const float* x   = (const float*)d_in[0];
  const float* cw0 = (const float*)d_in[1];  const float* cb0 = (const float*)d_in[2];
  const float* cw1 = (const float*)d_in[3];  const float* cb1 = (const float*)d_in[4];
  const float* cw2 = (const float*)d_in[5];  const float* cb2 = (const float*)d_in[6];
  const float* cw3 = (const float*)d_in[7];  const float* cb3 = (const float*)d_in[8];
  const float* cw4 = (const float*)d_in[9];  const float* cb4 = (const float*)d_in[10];
  const float* clw = (const float*)d_in[11]; const float* clb = (const float*)d_in[12];
  const float* hw  = (const float*)d_in[13]; const float* hb  = (const float*)d_in[14];
  const float* ha  = (const float*)d_in[15];
  const float* b0w = (const float*)d_in[16]; const float* b0b = (const float*)d_in[17];
  const float* mw  = (const float*)d_in[18]; const float* mbp = (const float*)d_in[19];
  const float* ba  = (const float*)d_in[20];
  const float* b6w = (const float*)d_in[21]; const float* b6b = (const float*)d_in[22];
  const float* tw  = (const float*)d_in[23]; const float* tb  = (const float*)d_in[24];

  half_t* ws = (half_t*)d_ws;
  int* route = (int*)((char*)d_ws + OFF_ROUTE_BYTES);

  prep_weights<<<dim3(128), dim3(256), 0, stream>>>(cw0, cw1, cw2, cw3, cw4,
                                                    hw, b0w, mw, b6w, ws);
  cls_kernel<<<dim3(512), dim3(256), SMEM_CLS, stream>>>(x, ws, cb0, cb1, cb2,
                                                         cb3, cb4, clw, clb, route);
  branch_kernel<<<dim3(512), dim3(256), SMEM_BR, stream>>>(x, ws, route, hb, ha,
                                                           b0b, mbp, ba, b6b, tw,
                                                           tb, (float*)d_out);
}
